// RLST_56298431315997
// MI455X (gfx1250) — compile-verified, matrix operand order checked by reading, not run on hardware
//
#include <hip/hip_runtime.h>
#include <hip/hip_bf16.h>

// ---------------------------------------------------------------------------
// RLST scan on MI455X (gfx1250):
//  - weights converted once per launch to bf16, transposed to N-major
//  - all GEMMs via v_wmma_f32_16x16x32_bf16 (f32 accumulate)
//  - B (weight) tiles staged into LDS once per block via the Tensor Data
//    Mover (tensor_load_to_lds + s_wait_tensorcnt); waves read B fragments
//    with ds_load_b128. A (activation) fragments stream from global/L2.
//  - 95 serial steps, each: bookkeep -> gates0 -> gru0 -> gates1 -> gru1 ->
//    out-projection -> finish-qt (host-enqueued chain, graph-captured)
// ---------------------------------------------------------------------------

#define BB       128
#define SS       48
#define TTT      48
#define TSTEPS   95        // S + Tt - 1
#define EE       256
#define HH       512
#define KK       512       // = 2E = H (both GEMM K dims)
#define GG       1536      // 3H
#define VV       10000
#define VQ       10002
#define VO       10016     // padded to /16
#define SRC_EOS_TOK  3
#define SRC_NULL_TOK 0
#define TRG_EOS_TOK  3
#define TRG_NULL_TOK 0
#define DISCOUNT_F   0.9f
#define PENALTY_M    7.0f

typedef __attribute__((ext_vector_type(16))) __bf16 v16bf;
typedef __attribute__((ext_vector_type(8)))  __bf16 bf16x8;
typedef __attribute__((ext_vector_type(8)))  float  v8f;
typedef __hip_bfloat16 bf16;

#if defined(__HIP_DEVICE_COMPILE__) && \
    __has_builtin(__builtin_amdgcn_tensor_load_to_lds) && \
    __has_builtin(__builtin_amdgcn_s_wait_tensorcnt)
#define HAVE_TDM 1
#else
#define HAVE_TDM 0
#endif

#if HAVE_TDM
typedef __attribute__((ext_vector_type(4))) unsigned int u32x4;
typedef __attribute__((ext_vector_type(8))) int          i32x8;
typedef __attribute__((ext_vector_type(4))) int          i32x4;

// TDM DMA of a (rows x k) bf16 tile (row stride = k) from global into LDS.
// D# built per CDNA5 ISA 8.3/8.4: group0 = {flags, lds_addr, global_addr,
// addr_hi|type=2}; group1 packs data_size=2B, tensor dims == tile dims,
// tensor_dim0_stride = k. Groups 2/3 zero (2D tensor).
__device__ __forceinline__ void tdm_load_2d(const bf16* __restrict__ gsrc,
                                            bf16* lds_dst, int rows, int k) {
  unsigned long long ga = (unsigned long long)(size_t)gsrc;
  unsigned lds = (unsigned)(size_t)lds_dst;   // LDS aperture: low 32b = offset
  u32x4 g0 = { 1u,                                    // count=1, user mode
               lds,                                   // lds_addr
               (unsigned)(ga & 0xFFFFFFFFu),          // global_addr[31:0]
               (unsigned)((ga >> 32) & 0x01FFFFFFu) | (2u << 30) }; // [56:32]|type=2
  i32x8 g1 = { (int)0x00010000u,                      // data_size=1 (2 bytes)
               (int)(((unsigned)k & 0xFFFFu) << 16),  // tensor_dim0[15:0]
               (int)((((unsigned)k >> 16) & 0xFFFFu) | (((unsigned)rows & 0xFFFFu) << 16)),
               (int)(((unsigned)k & 0xFFFFu) << 16),  // tile_dim0 = k
               rows,                                  // tile_dim1 = rows
               k,                                     // tensor_dim0_stride[31:0]
               0, 0 };
  i32x4 z4 = { 0, 0, 0, 0 };
#if __clang_major__ >= 23
  i32x8 z8 = { 0, 0, 0, 0, 0, 0, 0, 0 };
  __builtin_amdgcn_tensor_load_to_lds(g0, g1, z4, z4, z8, 0);
#else
  __builtin_amdgcn_tensor_load_to_lds(g0, g1, z4, z4, 0);
#endif
}
#endif

// Stage a weight tile into LDS: TDM path (one DMA per block, wave 0 issues)
// or cooperative-copy fallback. Caller must run stage_fence() before use.
__device__ __forceinline__ void stage_tile(const bf16* __restrict__ g,
                                           bf16* s, int rows) {
#if HAVE_TDM
  if (threadIdx.x < 32) tdm_load_2d(g, s, rows, KK);
#else
  const uint4* gs = (const uint4*)g;
  uint4* ss = (uint4*)s;
  int nv = rows * KK / 8;                    // 8 bf16 per 16B
  for (int i = threadIdx.x; i < nv; i += blockDim.x) ss[i] = gs[i];
#endif
}

__device__ __forceinline__ void stage_fence() {
#if HAVE_TDM
  if (threadIdx.x < 32) __builtin_amdgcn_s_wait_tensorcnt(0);
#endif
  __syncthreads();
}

// --- WMMA fragment load (works for global or LDS pointers after inlining).
// Per CDNA5 16-bit A/B layout: lanes 0-15 -> K {0..7, 16..23},
// lanes 16-31 -> K {8..15, 24..31}; each chunk is 8 bf16 = 16 bytes.
__device__ __forceinline__ v16bf load_frag(const bf16* base, int ld,
                                           int row0, int k0) {
  int lane = threadIdx.x & 31;
  int r    = row0 + (lane & 15);
  int kadd = (lane < 16) ? 0 : 8;
  const bf16* p = base + (size_t)r * ld + k0 + kadd;
  union { v16bf v; bf16x8 h[2]; } u;
  u.h[0] = *(const bf16x8*)(p);
  u.h[1] = *(const bf16x8*)(p + 16);
  return u.v;
}

// --- Dual GEMM for GRU gates: O1 = A1 @ W1^T, O2 = A2 @ W2^T
// Block: one N tile (both weight tiles TDM-staged in LDS, 32 KB), 8 waves
// cover the 8 M tiles; each wave runs two WMMA chains (latency hiding).
__global__ void __launch_bounds__(256)
k_gemm_gates(const bf16* __restrict__ A1, const bf16* __restrict__ A2,
             const bf16* __restrict__ W1, const bf16* __restrict__ W2,
             float* __restrict__ O1, float* __restrict__ O2) {
  __shared__ __align__(16) bf16 Bs1[16 * KK];
  __shared__ __align__(16) bf16 Bs2[16 * KK];
  int wave = threadIdx.x >> 5;
  int lane = threadIdx.x & 31;
  int n = blockIdx.x;            // 96 N tiles
  int m = wave;                  // 8 M tiles
  stage_tile(W1 + (size_t)n * 16 * KK, Bs1, 16);
  stage_tile(W2 + (size_t)n * 16 * KK, Bs2, 16);
  stage_fence();

  int row0 = m * 16, col0 = n * 16;
  v8f acc1 = {}; v8f acc2 = {};
#pragma unroll 4
  for (int kb = 0; kb < KK; kb += 32) {
    v16bf a1 = load_frag(A1, KK, row0, kb);
    v16bf b1 = load_frag(Bs1, KK, 0, kb);
    v16bf a2 = load_frag(A2, KK, row0, kb);
    v16bf b2 = load_frag(Bs2, KK, 0, kb);
    acc1 = __builtin_amdgcn_wmma_f32_16x16x32_bf16(false, a1, false, b1,
                                                   (short)0, acc1, false, false);
    acc2 = __builtin_amdgcn_wmma_f32_16x16x32_bf16(false, a2, false, b2,
                                                   (short)0, acc2, false, false);
  }
  int col   = col0 + (lane & 15);
  int rbase = row0 + ((lane < 16) ? 0 : 8);
#pragma unroll
  for (int r = 0; r < 8; ++r) {
    O1[(size_t)(rbase + r) * GG + col] = acc1[r];
    O2[(size_t)(rbase + r) * GG + col] = acc2[r];
  }
}

// --- Output projection: logits = h1 @ Wout^T + b_out
// Block: one N tile of Wout (TDM-staged, 16 KB), 8 waves = 8 M tiles.
__global__ void __launch_bounds__(256)
k_gemm_out(const bf16* __restrict__ A, const bf16* __restrict__ W,
           const float* __restrict__ bias, float* __restrict__ O) {
  __shared__ __align__(16) bf16 Bs[16 * KK];
  int wave = threadIdx.x >> 5;
  int lane = threadIdx.x & 31;
  int n = blockIdx.x;            // 626 N tiles
  int m = wave;                  // 8 M tiles
  stage_tile(W + (size_t)n * 16 * KK, Bs, 16);
  if (threadIdx.x == 32 && n + 1 < VO / 16)       // hint next tile toward L2
    __builtin_prefetch(W + (size_t)(n + 1) * 16 * KK, 0, 1);
  stage_fence();

  int row0 = m * 16, col0 = n * 16;
  v8f acc = {};
#pragma unroll 4
  for (int kb = 0; kb < KK; kb += 32) {
    v16bf a = load_frag(A, KK, row0, kb);
    v16bf b = load_frag(Bs, KK, 0, kb);
    acc = __builtin_amdgcn_wmma_f32_16x16x32_bf16(false, a, false, b,
                                                  (short)0, acc, false, false);
  }
  int col   = col0 + (lane & 15);
  float bv  = (col < VQ) ? bias[col] : 0.0f;
  int rbase = row0 + ((lane < 16) ? 0 : 8);
#pragma unroll
  for (int r = 0; r < 8; ++r)
    O[(size_t)(rbase + r) * VO + col] = acc[r] + bv;
}

// --- GRU elementwise: h = (1-z)*n + z*h   (in-place on bf16 h)
__global__ void __launch_bounds__(256)
k_gru(const float* __restrict__ gi, const float* __restrict__ gh,
      const float* __restrict__ bih, const float* __restrict__ bhh,
      bf16* __restrict__ h) {
  int idx = blockIdx.x * 256 + threadIdx.x;   // 128*512 = 65536
  int b = idx >> 9, c = idx & 511;
  const float* gib = gi + (size_t)b * GG;
  const float* ghb = gh + (size_t)b * GG;
  float ir = gib[c]        + bih[c],        hr = ghb[c]        + bhh[c];
  float iz = gib[c + HH]   + bih[c + HH],   hz = ghb[c + HH]   + bhh[c + HH];
  float in_= gib[c + 2*HH] + bih[c + 2*HH], hn = ghb[c + 2*HH] + bhh[c + 2*HH];
  float r = 1.f / (1.f + __expf(-(ir + hr)));
  float z = 1.f / (1.f + __expf(-(iz + hz)));
  float nn = tanhf(in_ + r * hn);
  float hp = __bfloat162float(h[idx]);
  h[idx] = __float2bfloat16((1.f - z) * nn + z * hp);
}

// --- Scan bookkeeping (consumes previous step's logits), per batch row.
__global__ void __launch_bounds__(256)
k_bookkeep(const float* __restrict__ logits, const int* __restrict__ src,
           const int* __restrict__ trg,
           int* ii, int* jj, int* term, const int* action,
           int* f_reading, int* f_naughty, int* f_justterm,
           float* reward_arr, int* counts,
           const bf16* __restrict__ Esrc, const bf16* __restrict__ Etrg,
           bf16* __restrict__ X,
           float* __restrict__ tok, float* __restrict__ qu, int t) {
  __shared__ float smax[256]; __shared__ int sidx[256]; __shared__ float ssum[256];
  __shared__ int sh_inp, sh_wout, sh_writing, sh_oldj;
  int b = blockIdx.x, tid = threadIdx.x;
  const float* lg = logits + (size_t)b * VO;

  // argmax + max over V (first index wins ties, like jnp.argmax)
  float mv = -3.4e38f; int mi = 0;
  for (int c = tid; c < VV; c += 256) { float v = lg[c]; if (v > mv) { mv = v; mi = c; } }
  smax[tid] = mv; sidx[tid] = mi; __syncthreads();
  for (int s = 128; s > 0; s >>= 1) {
    if (tid < s) {
      float ov = smax[tid + s]; int oi = sidx[tid + s];
      if (ov > smax[tid] || (ov == smax[tid] && oi < sidx[tid])) { smax[tid] = ov; sidx[tid] = oi; }
    }
    __syncthreads();
  }
  float gmax = smax[0]; int wout_am = sidx[0];
  float se = 0.f;
  for (int c = tid; c < VV; c += 256) se += __expf(lg[c] - gmax);
  ssum[tid] = se; __syncthreads();
  for (int s = 128; s > 0; s >>= 1) { if (tid < s) ssum[tid] += ssum[tid + s]; __syncthreads(); }
  float lse = gmax + __logf(ssum[0]);

  if (tid == 0) {
    int t0 = term[b], act = action[b], i0 = ii[b], j0 = jj[b];
    float q_used  = t0 ? 0.f : lg[VV + act];
    int reading   = (!t0) && (act == 0);
    int writing   = (!t0) && (act == 1);
    int just_term = writing && (trg[b * TTT + j0] == TRG_EOS_TOK);
    int naughty   = reading && (src[b * SS + i0] == SRC_EOS_TOK);
    int i1 = i0 + ((!naughty && reading) ? 1 : 0); if (i1 > SS - 1) i1 = SS - 1;
    int old_j = j0;
    int j1 = j0 + ((writing && !just_term) ? 1 : 0); if (j1 > TTT - 1) j1 = TTT - 1;
    int t1 = t0 || just_term;
    int wout_tok = reading ? TRG_NULL_TOK : wout_am;
    int tgt = trg[b * TTT + old_j];
    float reward = lg[tgt] - lse;
    int inp = src[b * SS + i1];
    if (writing) inp = SRC_NULL_TOK;
    if (naughty) inp = SRC_EOS_TOK;
    if (reading) atomicAdd(&counts[0], 1);
    if (writing) atomicAdd(&counts[1], 1);
    ii[b] = i1; jj[b] = j1; term[b] = t1;
    f_reading[b] = reading; f_naughty[b] = naughty; f_justterm[b] = just_term;
    reward_arr[b] = reward;
    qu[(size_t)b * TSTEPS + t] = q_used;
    sh_inp = inp; sh_wout = wout_tok; sh_writing = writing; sh_oldj = old_j;
  }
  __syncthreads();
  if (sh_writing) {
    float* trow = tok + ((size_t)b * TTT + sh_oldj) * VV;
    for (int c = tid; c < VV; c += 256) trow[c] += lg[c];
  }
  // build next-step X = [emb_src[inp], emb_trg[wout]]
  X[(size_t)b * KK + tid]      = Esrc[(size_t)sh_inp  * EE + tid];
  X[(size_t)b * KK + EE + tid] = Etrg[(size_t)sh_wout * EE + tid];
}

// --- After new logits: compute nbav/action and finish qt for step t.
__global__ void k_finish(const float* __restrict__ logits, const int* term,
                         const int* f_reading, const int* f_naughty,
                         const int* f_justterm, const float* reward_arr,
                         int* action, float* __restrict__ qt_out, int t) {
  int b = blockIdx.x * blockDim.x + threadIdx.x;
  if (b >= BB) return;
  float q0 = logits[(size_t)b * VO + VV];
  float q1 = logits[(size_t)b * VO + VV + 1];
  float nbav = fmaxf(q0, q1);
  action[b] = (q1 > q0) ? 1 : 0;
  float reward = reward_arr[b];
  float qt = reward + DISCOUNT_F * nbav;
  if (term[b])       qt = 0.f;
  if (f_reading[b])  qt = nbav;
  if (f_naughty[b])  qt = DISCOUNT_F * nbav;
  if (f_justterm[b]) qt = reward;
  if (f_naughty[b])  qt -= PENALTY_M;
  qt_out[(size_t)b * TSTEPS + t] = qt;
}

__global__ void k_init_action(const float* __restrict__ logits, int* action) {
  int b = threadIdx.x;
  if (b >= BB) return;
  float q0 = logits[(size_t)b * VO + VV];
  float q1 = logits[(size_t)b * VO + VV + 1];
  action[b] = (q1 > q0) ? 1 : 0;
}

// --- One-time per-launch conversions -------------------------------------
__global__ void k_convert_tr(const float* __restrict__ W, bf16* __restrict__ Wt,
                             int K, int N, int Npad) {
  size_t total = (size_t)Npad * K;
  for (size_t idx = (size_t)blockIdx.x * 256 + threadIdx.x; idx < total;
       idx += (size_t)gridDim.x * 256) {
    int n = (int)(idx / K), k = (int)(idx % K);
    float v = (n < N) ? W[(size_t)k * N + n] : 0.f;
    Wt[idx] = __float2bfloat16(v);
  }
}

__global__ void k_convert_emb(const float* __restrict__ E, bf16* __restrict__ Eb,
                              size_t n) {
  for (size_t idx = (size_t)blockIdx.x * 256 + threadIdx.x; idx < n;
       idx += (size_t)gridDim.x * 256)
    Eb[idx] = __float2bfloat16(E[idx]);
}

__global__ void k_zero(float* __restrict__ p, size_t n) {
  for (size_t i = (size_t)blockIdx.x * 256 + threadIdx.x; i < n;
       i += (size_t)gridDim.x * 256)
    p[i] = 0.f;
}

__global__ void k_init(const int* __restrict__ src, const bf16* __restrict__ Esrc,
                       const bf16* __restrict__ Etrg, bf16* __restrict__ X,
                       bf16* __restrict__ h0, bf16* __restrict__ h1,
                       int* ii, int* jj, int* term, int* counts) {
  int b = blockIdx.x, tid = threadIdx.x;
  if (b == 0 && tid < 2) counts[tid] = 0;
  if (tid == 0) { ii[b] = 0; jj[b] = 0; term[b] = 0; }
  int inp = src[b * SS];
  X[(size_t)b * KK + tid]      = Esrc[(size_t)inp * EE + tid];
  X[(size_t)b * KK + EE + tid] = Etrg[(size_t)TRG_NULL_TOK * EE + tid];
  bf16 z = __float2bfloat16(0.f);
  h0[(size_t)b * HH + tid] = z; h0[(size_t)b * HH + 256 + tid] = z;
  h1[(size_t)b * HH + tid] = z; h1[(size_t)b * HH + 256 + tid] = z;
}

__global__ void k_counts_out(const int* __restrict__ counts, float* __restrict__ o) {
  if (threadIdx.x < 2) o[threadIdx.x] = (float)counts[threadIdx.x];
}

// ---------------------------------------------------------------------------
extern "C" void kernel_launch(void* const* d_in, const int* in_sizes, int n_in,
                              void* d_out, int out_size, void* d_ws, size_t ws_size,
                              hipStream_t stream) {
  const int*   src     = (const int*)d_in[0];
  const int*   trg     = (const int*)d_in[1];
  const float* emb_src = (const float*)d_in[2];
  const float* emb_trg = (const float*)d_in[3];
  const float* w_ih_0  = (const float*)d_in[4];
  const float* w_hh_0  = (const float*)d_in[5];
  const float* b_ih_0  = (const float*)d_in[6];
  const float* b_hh_0  = (const float*)d_in[7];
  const float* w_ih_1  = (const float*)d_in[8];
  const float* w_hh_1  = (const float*)d_in[9];
  const float* b_ih_1  = (const float*)d_in[10];
  const float* b_hh_1  = (const float*)d_in[11];
  const float* w_out   = (const float*)d_in[12];
  const float* b_out   = (const float*)d_in[13];

  char* wp = (char*)d_ws;
  auto carve = [&](size_t bytes) -> void* {
    void* p = (void*)wp; wp += (bytes + 255) & ~(size_t)255; return p;
  };
  bf16*  wt_ih0 = (bf16*)carve((size_t)GG * KK * 2);
  bf16*  wt_hh0 = (bf16*)carve((size_t)GG * KK * 2);
  bf16*  wt_ih1 = (bf16*)carve((size_t)GG * KK * 2);
  bf16*  wt_hh1 = (bf16*)carve((size_t)GG * KK * 2);
  bf16*  wt_out = (bf16*)carve((size_t)VO * KK * 2);
  bf16*  Esrc   = (bf16*)carve((size_t)16000 * EE * 2);
  bf16*  Etrg   = (bf16*)carve((size_t)VV * EE * 2);
  bf16*  X      = (bf16*)carve((size_t)BB * KK * 2);
  bf16*  h0     = (bf16*)carve((size_t)BB * HH * 2);
  bf16*  h1     = (bf16*)carve((size_t)BB * HH * 2);
  float* gi0    = (float*)carve((size_t)BB * GG * 4);
  float* gh0    = (float*)carve((size_t)BB * GG * 4);
  float* gi1    = (float*)carve((size_t)BB * GG * 4);
  float* gh1    = (float*)carve((size_t)BB * GG * 4);
  float* logits = (float*)carve((size_t)BB * VO * 4);
  int*   ii     = (int*)carve(BB * 4);
  int*   jj     = (int*)carve(BB * 4);
  int*   term   = (int*)carve(BB * 4);
  int*   action = (int*)carve(BB * 4);
  int*   f_read = (int*)carve(BB * 4);
  int*   f_ngt  = (int*)carve(BB * 4);
  int*   f_jt   = (int*)carve(BB * 4);
  float* reward = (float*)carve(BB * 4);
  int*   counts = (int*)carve(2 * 4);

  float* out_f = (float*)d_out;
  float* tok = out_f;                                   // B*Tt*V
  float* qu  = tok + (size_t)BB * TTT * VV;             // B*T
  float* qt  = qu + (size_t)BB * TSTEPS;                // B*T
  float* cnt = qt + (size_t)BB * TSTEPS;                // 2

  // one-time conversions + init
  k_convert_tr<<<2048, 256, 0, stream>>>(w_ih_0, wt_ih0, KK, GG, GG);
  k_convert_tr<<<2048, 256, 0, stream>>>(w_hh_0, wt_hh0, KK, GG, GG);
  k_convert_tr<<<2048, 256, 0, stream>>>(w_ih_1, wt_ih1, KK, GG, GG);
  k_convert_tr<<<2048, 256, 0, stream>>>(w_hh_1, wt_hh1, KK, GG, GG);
  k_convert_tr<<<2048, 256, 0, stream>>>(w_out, wt_out, KK, VQ, VO);
  k_convert_emb<<<2048, 256, 0, stream>>>(emb_src, Esrc, (size_t)16000 * EE);
  k_convert_emb<<<2048, 256, 0, stream>>>(emb_trg, Etrg, (size_t)VV * EE);
  k_zero<<<2048, 256, 0, stream>>>(out_f, (size_t)out_size);
  k_init<<<BB, 256, 0, stream>>>(src, Esrc, Etrg, X, h0, h1, ii, jj, term, counts);

  auto approx = [&]() {
    k_gemm_gates<<<GG / 16, 256, 0, stream>>>(X, h0, wt_ih0, wt_hh0, gi0, gh0);
    k_gru<<<256, 256, 0, stream>>>(gi0, gh0, b_ih_0, b_hh_0, h0);
    k_gemm_gates<<<GG / 16, 256, 0, stream>>>(h0, h1, wt_ih1, wt_hh1, gi1, gh1);
    k_gru<<<256, 256, 0, stream>>>(gi1, gh1, b_ih_1, b_hh_1, h1);
    k_gemm_out<<<VO / 16, 256, 0, stream>>>(h1, wt_out, b_out, logits);
  };

  approx();                                   // out0 / state0
  k_init_action<<<1, 128, 0, stream>>>(logits, action);

  for (int t = 0; t < TSTEPS; ++t) {
    k_bookkeep<<<BB, 256, 0, stream>>>(logits, src, trg, ii, jj, term, action,
                                       f_read, f_ngt, f_jt, reward, counts,
                                       Esrc, Etrg, X, tok, qu, t);
    approx();                                 // next out / state
    k_finish<<<1, 128, 0, stream>>>(logits, term, f_read, f_ngt, f_jt,
                                    reward, action, qt, t);
  }
  k_counts_out<<<1, 2, 0, stream>>>(counts, cnt);
}
